// RobertaSelfAttention_matchKV_49340584296640
// MI455X (gfx1250) — matrix operand order repaired and checked by reading, not compiled, not run on hardware
//
#include <hip/hip_runtime.h>
#include <hip/hip_bf16.h>

typedef __attribute__((ext_vector_type(16))) _Float16 v16h;
typedef __attribute__((ext_vector_type(8)))  _Float16 v8h;
typedef __attribute__((ext_vector_type(8)))  float    v8f;

#define BB   4
#define LL   2048
#define HIDD 768
#define NH   12
#define HD   64
#define NR   4
#define MM   (BB * LL)      // 8192
#define KK   HIDD           // 768
#define NN   HIDD           // 768

// ---------------------------------------------------------------------------
// gfx1250 async global->LDS copy (ASYNCcnt-tracked) + wait
// ---------------------------------------------------------------------------
__device__ __forceinline__ void async_copy_b128(void* lds_dst, const void* gsrc) {
    // low 32 bits of the generic (flat) shared pointer == wave-relative LDS offset
    uint32_t ldsoff = (uint32_t)(uintptr_t)lds_dst;
    asm volatile("global_load_async_to_lds_b128 %0, %1, off"
                 :: "v"(ldsoff), "v"(gsrc) : "memory");
}
__device__ __forceinline__ void wait_async0() {
    asm volatile("s_wait_asynccnt 0x0" ::: "memory");
}

__device__ __forceinline__ v16h cat8(v8h lo, v8h hi) {
    return __builtin_shufflevector(lo, hi, 0, 1, 2, 3, 4, 5, 6, 7,
                                           8, 9, 10, 11, 12, 13, 14, 15);
}

// stage one 64x32-half A slab and one 64x32-half W slab into LDS (async)
__device__ __forceinline__ void stage_tiles(const _Float16* __restrict__ A,
                                            const _Float16* __restrict__ W,
                                            _Float16 (*As)[32], _Float16 (*Ws)[32],
                                            int m_base, int n_base, int K, int k0,
                                            int tid) {
#pragma unroll
    for (int c = tid; c < 256; c += 128) {   // 256 16-byte chunks per slab
        const int row = c >> 2;
        const int off = (c & 3) * 8;
        async_copy_b128(&As[row][off], A + (long)(m_base + row) * K + k0 + off);
        async_copy_b128(&Ws[row][off], W + (long)(n_base + row) * K + k0 + off);
    }
}

// ---------------------------------------------------------------------------
// C = relu(A(M,K) * W(N,K)^T + bias) ; A,W pre-converted to f16.
// 128-thread block = 4 waves -> 64x64 tile; each wave: 16(M) x 64(N), 4 WMMAs
// per 32-wide K step, LDS double-buffered via async copies.
// ---------------------------------------------------------------------------
template <typename OutT>
__global__ void __launch_bounds__(128)
wmma_gemm_bias_relu(const _Float16* __restrict__ A, const _Float16* __restrict__ Wt,
                    const float* __restrict__ bias, OutT* __restrict__ C,
                    int K, int N) {
    __shared__ __align__(16) _Float16 As[2][64][32];
    __shared__ __align__(16) _Float16 Ws[2][64][32];

    const int tid    = threadIdx.x;
    const int lane   = tid & 31;
    const int wave   = tid >> 5;              // M sub-tile 0..3
    const int m_base = blockIdx.y * 64;
    const int n_base = blockIdx.x * 64;

    v8f acc[4] = {v8f{}, v8f{}, v8f{}, v8f{}};

    stage_tiles(A, Wt, As[0], Ws[0], m_base, n_base, K, 0, tid);

    const int nsteps = K / 32;
    for (int s = 0; s < nsteps; ++s) {
        wait_async0();
        __syncthreads();                      // slab s resident for all waves
        if (s + 1 < nsteps)
            stage_tiles(A, Wt, As[(s + 1) & 1], Ws[(s + 1) & 1],
                        m_base, n_base, K, (s + 1) * 32, tid);

        const int buf = s & 1;
        // A fragment (16x32 f16 layout): rows M=lane&15 in both halves,
        // lane-half picks K sub-stripe {0..7,16..23} vs {8..15,24..31}
        const int arow = wave * 16 + (lane & 15);
        const int kb   = ((lane >> 4) & 1) * 8;
        const v16h afrag = cat8(*(const v8h*)&As[buf][arow][kb],
                                *(const v8h*)&As[buf][arow][16 + kb]);

        // B fragment (32x16): col n=lane&15; lane-half picks K {0..7,16..23} vs {8..15,24..31}
        const int koff = ((lane >> 4) & 1) * 8;
#pragma unroll
        for (int nt = 0; nt < 4; ++nt) {
            const int brow = nt * 16 + (lane & 15);
            const v16h bfrag = cat8(*(const v8h*)&Ws[buf][brow][koff],
                                    *(const v8h*)&Ws[buf][brow][koff + 16]);
            acc[nt] = __builtin_amdgcn_wmma_f32_16x16x32_f16(
                false, afrag, false, bfrag, (short)0, acc[nt], false, false);
        }
        __syncthreads();                      // reads done before overwrite
    }

    // C/D layout: row = (lane>=16)*8 + j, col = lane&15
    const int col16 = lane & 15;
    const int rbase = m_base + wave * 16 + ((lane >> 4) & 1) * 8;
#pragma unroll
    for (int nt = 0; nt < 4; ++nt) {
        const int col  = n_base + nt * 16 + col16;
        const float bv = bias[col];
#pragma unroll
        for (int j = 0; j < 8; ++j) {
            float v = acc[nt][j] + bv;
            C[(long)(rbase + j) * N + col] = (OutT)(v > 0.f ? v : 0.f);
        }
    }
}

// elementwise f32 -> f16
__global__ void cvt_f32_f16(const float* __restrict__ in, _Float16* __restrict__ out,
                            long n) {
    const long i = (long)blockIdx.x * blockDim.x + threadIdx.x;
    if (i < n) out[i] = (_Float16)in[i];
}

// dot[bl,h] = sum_d K1[bl, h*64+d] * reading_head[h,d]
__global__ void dot_kernel(const _Float16* __restrict__ K1,
                           const float* __restrict__ rh,
                           float* __restrict__ dot) {
    const int idx = blockIdx.x * blockDim.x + threadIdx.x;
    if (idx >= MM * NH) return;
    const int h  = idx % NH;
    const int bl = idx / NH;
    const _Float16* kp = K1 + (long)bl * HIDD + h * HD;
    const float* rp = rh + h * HD;
    float s = 0.f;
#pragma unroll 8
    for (int d = 0; d < HD; ++d) s += (float)kp[d] * rp[d];
    dot[idx] = s;
}

// per-(b,h): softmax-variant over L -> valid mask
__global__ void __launch_bounds__(256)
valid_kernel(const float* __restrict__ dot, int* __restrict__ valid, float thresh) {
    const int b = blockIdx.x / NH;
    const int h = blockIdx.x % NH;
    __shared__ float red[256];
    const int tid = threadIdx.x;

    float mx = -3.4e38f;
    for (int l = tid; l < LL; l += 256)
        mx = fmaxf(mx, dot[(long)(b * LL + l) * NH + h]);
    red[tid] = mx;
    __syncthreads();
    for (int s = 128; s > 0; s >>= 1) {
        if (tid < s) red[tid] = fmaxf(red[tid], red[tid + s]);
        __syncthreads();
    }
    const float gmax = red[0];
    __syncthreads();

    float sum = 0.f;
    for (int l = tid; l < LL; l += 256)
        sum += __expf(dot[(long)(b * LL + l) * NH + h] - gmax);
    red[tid] = sum;
    __syncthreads();
    for (int s = 128; s > 0; s >>= 1) {
        if (tid < s) red[tid] += red[tid + s];
        __syncthreads();
    }
    const float gsum   = red[0];
    const float addend = __expf(-gmax);

    for (int l = tid; l < LL; l += 256) {
        float sm = __expf(dot[(long)(b * LL + l) * NH + h] - gmax) / gsum + addend;
        valid[(long)(b * LL + l) * NH + h] = (sm > thresh) ? 1 : 0;
    }
}

__device__ __forceinline__ void store_regs(int* __restrict__ dst, int b, int l,
                                           const int* regs, int h) {
#pragma unroll
    for (int r = 0; r < NR; ++r)
        dst[(((long)b * LL + l) * NR + r) * NH + h] = regs[r];
}

// sequential register-shift scans; one thread per (b,h) chain
__global__ void scan_kernel(const int* __restrict__ valid,
                            int* __restrict__ fwd, int* __restrict__ bwd) {
    const int tid = blockIdx.x * blockDim.x + threadIdx.x;
    if (tid >= BB * NH) return;
    const int b = tid / NH;
    const int h = tid % NH;

    int regs[NR] = {0, 0, 0, 0};
    store_regs(fwd, b, 0, regs, h);
    for (int t = 1; t < LL; ++t) {
        if (valid[(long)(b * LL + t) * NH + h]) {
            regs[3] = regs[2]; regs[2] = regs[1]; regs[1] = regs[0]; regs[0] = t;
        }
        store_regs(fwd, b, t, regs, h);
    }

    int r2[NR] = {0, 0, 0, 0};
    r2[0] = valid[(long)(b * LL + (LL - 1)) * NH + h] ? (LL - 1) : 0;
    store_regs(bwd, b, LL - 1, r2, h);
    for (int t = LL - 2; t >= 1; --t) {
        if (valid[(long)(b * LL + t) * NH + h]) {
            r2[3] = r2[2]; r2[2] = r2[1]; r2[1] = r2[0]; r2[0] = t + 1;
        }
        store_regs(bwd, b, t, r2, h);
    }
    int z[NR] = {0, 0, 0, 0};
    store_regs(bwd, b, 0, z, h);
}

// new_states[b,l,h,d] = sum_r w[h,r]*V1[b,fwd,h,d] + w[h,R+r]*V1[b,bwd,h,d]
__global__ void gather_kernel(const _Float16* __restrict__ V1,
                              const int* __restrict__ fwd,
                              const int* __restrict__ bwd,
                              const float* __restrict__ w,
                              _Float16* __restrict__ ns) {
    const long idx = (long)blockIdx.x * blockDim.x + threadIdx.x;
    if (idx >= (long)MM * NH * HD) return;
    const int d   = (int)(idx & (HD - 1));
    const long t  = idx >> 6;
    const int h   = (int)(t % NH);
    const long bl = t / NH;
    const int b = (int)(bl / LL);
    const int l = (int)(bl % LL);

    float acc = 0.f;
#pragma unroll
    for (int r = 0; r < NR; ++r) {
        const long base = (((long)b * LL + l) * NR + r) * NH + h;
        const int fi = fwd[base];
        const int bi = bwd[base];
        acc += w[h * 2 * NR + r]      * (float)V1[((long)b * LL + fi) * HIDD + h * HD + d];
        acc += w[h * 2 * NR + NR + r] * (float)V1[((long)b * LL + bi) * HIDD + h * HD + d];
    }
    ns[bl * HIDD + h * HD + d] = (_Float16)acc;
}

extern "C" void kernel_launch(void* const* d_in, const int* in_sizes, int n_in,
                              void* d_out, int out_size, void* d_ws, size_t ws_size,
                              hipStream_t stream) {
    const float* hidden = (const float*)d_in[0];
    const float* K1_w   = (const float*)d_in[1];
    const float* K1_b   = (const float*)d_in[2];
    const float* V1_w   = (const float*)d_in[3];
    const float* V1_b   = (const float*)d_in[4];
    const float* P1_w   = (const float*)d_in[5];
    const float* P1_b   = (const float*)d_in[6];
    const float* rhead  = (const float*)d_in[7];
    const float* bidir  = (const float*)d_in[8];
    float* out = (float*)d_out;

    // ---- workspace carve-out ----
    char* p = (char*)d_ws;
    const size_t nAct = (size_t)MM * HIDD;     // 6,291,456
    const size_t nWgt = (size_t)NN * KK;       // 589,824
    _Float16* Ah   = (_Float16*)p; p += nAct * 2;
    _Float16* K1wh = (_Float16*)p; p += nWgt * 2;
    _Float16* V1wh = (_Float16*)p; p += nWgt * 2;
    _Float16* P1wh = (_Float16*)p; p += nWgt * 2;
    _Float16* K1h  = (_Float16*)p; p += nAct * 2;
    _Float16* V1h  = (_Float16*)p; p += nAct * 2;
    _Float16* nsh  = (_Float16*)p; p += nAct * 2;
    float* dotb = (float*)p;       p += (size_t)MM * NH * 4;
    int* valid  = (int*)p;         p += (size_t)MM * NH * 4;
    int* fwd    = (int*)p;         p += (size_t)MM * NR * NH * 4;
    int* bwd    = (int*)p;

    // ---- one-time f32 -> f16 conversions ----
    cvt_f32_f16<<<(int)((nAct + 255) / 256), 256, 0, stream>>>(hidden, Ah, (long)nAct);
    cvt_f32_f16<<<(int)((nWgt + 255) / 256), 256, 0, stream>>>(K1_w, K1wh, (long)nWgt);
    cvt_f32_f16<<<(int)((nWgt + 255) / 256), 256, 0, stream>>>(V1_w, V1wh, (long)nWgt);
    cvt_f32_f16<<<(int)((nWgt + 255) / 256), 256, 0, stream>>>(P1_w, P1wh, (long)nWgt);

    const dim3 gemmGrid(NN / 64, MM / 64);
    const dim3 gemmBlk(128);

    // K1 = relu(hidden @ K1_w^T + b), V1 likewise (f16 outputs)
    wmma_gemm_bias_relu<_Float16><<<gemmGrid, gemmBlk, 0, stream>>>(Ah, K1wh, K1_b, K1h, KK, NN);
    wmma_gemm_bias_relu<_Float16><<<gemmGrid, gemmBlk, 0, stream>>>(Ah, V1wh, V1_b, V1h, KK, NN);

    // reading-head dot + softmax-variant valid mask
    dot_kernel<<<(MM * NH + 255) / 256, 256, 0, stream>>>(K1h, rhead, dotb);
    valid_kernel<<<BB * NH, 256, 0, stream>>>(dotb, valid, 1.5f / (float)LL);

    // sequential register scans (48 chains)
    scan_kernel<<<1, 64, 0, stream>>>(valid, fwd, bwd);

    // irregular gather + bidirectional weighting (f16 in/out)
    const long gth = (long)MM * NH * HD;
    gather_kernel<<<(int)((gth + 255) / 256), 256, 0, stream>>>(V1h, fwd, bwd, bidir, nsh);

    // out = relu(ns @ P1_w^T + P1_b), f32 output
    wmma_gemm_bias_relu<float><<<gemmGrid, gemmBlk, 0, stream>>>(nsh, P1wh, P1_b, out, KK, NN);
}
